// AttnBlock_61684320305872
// MI455X (gfx1250) — matrix operand order repaired and checked by reading, not compiled, not run on hardware
//
#include <hip/hip_runtime.h>
#include <hip/hip_bf16.h>

typedef __attribute__((ext_vector_type(16))) _Float16 v16h;
typedef __attribute__((ext_vector_type(8)))  _Float16 h8;
typedef __attribute__((ext_vector_type(8)))  float    v8f;
typedef __attribute__((ext_vector_type(4)))  unsigned int u32x4;
typedef __attribute__((ext_vector_type(8)))  int          i32x8;
typedef __attribute__((ext_vector_type(4)))  int          i32x4;

union V16 { v16h v; h8 h[2]; };

#define C_DIM 512
#define HW    4096
#define BATCH 4
#define NG    32
#define CPG   16

static __device__ __forceinline__ v8f v8f_zero() {
  v8f v;
  #pragma unroll
  for (int i = 0; i < 8; ++i) v[i] = 0.0f;
  return v;
}

static __device__ __forceinline__ v8f wmma_f16(v16h a, v16h b, v8f c) {
  return __builtin_amdgcn_wmma_f32_16x16x32_f16(false, a, false, b, (short)0, c,
                                                false, false);
}

// ---------------------------------------------------------------------------
// TDM: 2-D tile load global->LDS via tensor descriptor (ISA 08 section 8).
// data_size = 2 bytes (f16). tile_d0 contiguous elements per row, tile_d1
// rows, row stride stride0 (elements). lds_off = byte offset in LDS.
// 6-arg builtin variant (clang-23 / therock HIP headers).
// ---------------------------------------------------------------------------
static __device__ __forceinline__ void tdm_load_2d_f16(unsigned lds_off,
                                                       const void* gptr,
                                                       unsigned tile_d0,
                                                       unsigned tile_d1,
                                                       unsigned long stride0) {
  unsigned long ga = (unsigned long)gptr;
  u32x4 g0;
  g0[0] = 1u;                                   // count=1, user descriptor
  g0[1] = lds_off;                              // lds_addr [63:32]
  g0[2] = (unsigned)(ga & 0xffffffffu);         // global_addr low
  g0[3] = (unsigned)((ga >> 32) & 0x01ffffffu)  // global_addr high (57-bit)
          | 0x80000000u;                        // type=2 ("image") [127:126]
  i32x8 g1;
  g1[0] = (int)(1u << 16);                      // data_size=1 -> 2 bytes
  g1[1] = (int)((tile_d0 & 0xffffu) << 16);     // tensor_dim0[15:0] @ bits63:48
  g1[2] = (int)((tile_d0 >> 16) |               // tensor_dim0[31:16]
                ((tile_d1 & 0xffffu) << 16));   // tensor_dim1[15:0]
  g1[3] = (int)((tile_d1 >> 16) |               // tensor_dim1[31:16]
                ((tile_d0 & 0xffffu) << 16));   // tile_dim0 @ bits127:112
  g1[4] = (int)(tile_d1 & 0xffffu);             // tile_dim1; tile_dim2=0
  g1[5] = (int)(stride0 & 0xffffffffu);         // tensor_dim0_stride low
  g1[6] = (int)((stride0 >> 32) & 0xffffu);     // tensor_dim0_stride high
  g1[7] = 0;                                    // tensor_dim1_stride (2-D)
  i32x4 z4;
  z4[0] = 0; z4[1] = 0; z4[2] = 0; z4[3] = 0;
  i32x8 z8;
  #pragma unroll
  for (int i = 0; i < 8; ++i) z8[i] = 0;
  __builtin_amdgcn_tensor_load_to_lds(g0, g1, z4, z4, z8, 0);
}

// ---------------------------------------------------------------------------
// Kernel 1: transpose weights [c][d] -> f16 [d][c]
// ---------------------------------------------------------------------------
__global__ void prep_w_kernel(const float* __restrict__ wq,
                              const float* __restrict__ wk,
                              const float* __restrict__ wv,
                              const float* __restrict__ wo,
                              _Float16* __restrict__ tq,
                              _Float16* __restrict__ tk,
                              _Float16* __restrict__ tv,
                              _Float16* __restrict__ to_) {
  int i = blockIdx.x * 256 + threadIdx.x;
  int w   = i >> 18;
  int rem = i & 262143;
  int c = rem >> 9;
  int d = rem & 511;
  const float* src = (w == 0) ? wq : (w == 1) ? wk : (w == 2) ? wv : wo;
  _Float16*    dst = (w == 0) ? tq : (w == 1) ? tk : (w == 2) ? tv : to_;
  dst[(size_t)d * C_DIM + c] = (_Float16)src[(size_t)c * C_DIM + d];
}

// ---------------------------------------------------------------------------
// Kernel 2: GroupNorm, write f16 token-major Xn[b][t][c]
// ---------------------------------------------------------------------------
__global__ void gn_kernel(const float* __restrict__ x,
                          const float* __restrict__ scale,
                          const float* __restrict__ bias,
                          _Float16* __restrict__ Xn) {
  int g = blockIdx.x;
  int b = blockIdx.y;
  int tid = threadIdx.x;
  const float* base = x + ((size_t)(b * C_DIM + g * CPG)) * HW;
  const int N = CPG * HW;
  float s = 0.0f, s2 = 0.0f;
  for (int i = tid; i < N; i += 256) {
    float v = base[i];
    s += v; s2 += v * v;
  }
  __shared__ float red[256], red2[256];
  red[tid] = s; red2[tid] = s2;
  __syncthreads();
  for (int off = 128; off > 0; off >>= 1) {
    if (tid < off) { red[tid] += red[tid + off]; red2[tid] += red2[tid + off]; }
    __syncthreads();
  }
  float mean = red[0] * (1.0f / N);
  float var  = red2[0] * (1.0f / N) - mean * mean;
  float rstd = rsqrtf(var + 1e-6f);
  for (int i = tid; i < N; i += 256) {
    int cl = i & (CPG - 1);
    int t  = i >> 4;
    int c  = g * CPG + cl;
    float v = base[(size_t)cl * HW + t];
    float y = (v - mean) * rstd * scale[c] + bias[c];
    Xn[((size_t)(b * HW + t)) * C_DIM + c] = (_Float16)y;
  }
}

// ---------------------------------------------------------------------------
// Kernel 3: QKV projections; weight tile (64x512) TDM-staged into LDS.
// ---------------------------------------------------------------------------
__global__ void qkv_kernel(const _Float16* __restrict__ Xn,
                           const _Float16* __restrict__ wqT,
                           const _Float16* __restrict__ wkT,
                           const _Float16* __restrict__ wvT,
                           const float* __restrict__ bq,
                           const float* __restrict__ bk,
                           const float* __restrict__ bv,
                           _Float16* __restrict__ Q,
                           _Float16* __restrict__ K,
                           _Float16* __restrict__ V) {
  extern __shared__ _Float16 Ws[];              // [64][512] f16 = 64 KB
  int z = blockIdx.z;
  int b = z / 3, which = z % 3;
  const _Float16* WT = (which == 0) ? wqT : (which == 1) ? wkT : wvT;
  const float* bias  = (which == 0) ? bq  : (which == 1) ? bk  : bv;

  int wave = threadIdx.x >> 5;
  int lane = threadIdx.x & 31;
  int lane15 = lane & 15;
  int halfId = lane >> 4;
  int m0 = blockIdx.y * 128 + wave * 16;
  int n0 = blockIdx.x * 64;

  if (threadIdx.x < 32) {
    tdm_load_2d_f16(0u, WT + (size_t)n0 * C_DIM, C_DIM, 64, C_DIM);
    __builtin_amdgcn_s_wait_tensorcnt(0);
  }
  __syncthreads();

  const h8* Arow = (const h8*)(Xn + ((size_t)(b * HW + m0 + lane15)) * C_DIM);
  v8f acc[4];
  #pragma unroll
  for (int j = 0; j < 4; ++j) acc[j] = v8f_zero();

  #pragma unroll
  for (int ks = 0; ks < 16; ++ks) {
    V16 a;
    a.h[0] = Arow[ks * 4 + halfId];
    a.h[1] = Arow[ks * 4 + halfId + 2];
    #pragma unroll
    for (int j = 0; j < 4; ++j) {
      const h8* Brow = (const h8*)(Ws + (size_t)(j * 16 + lane15) * C_DIM);
      V16 bb;
      bb.h[0] = Brow[ks * 4 + halfId];
      bb.h[1] = Brow[ks * 4 + halfId + 2];
      acc[j] = wmma_f16(a.v, bb.v, acc[j]);
    }
  }

  const float qscale = 0.044194173824159216f;   // 512^-0.5 folded into Q
  #pragma unroll
  for (int j = 0; j < 4; ++j) {
    int d = n0 + j * 16 + lane15;
    float bsv = bias[d];
    #pragma unroll
    for (int r = 0; r < 8; ++r) {
      int row = m0 + r + 8 * halfId;
      float val = acc[j][r] + bsv;
      if (which == 0)
        Q[((size_t)(b * HW + row)) * C_DIM + d] = (_Float16)(val * qscale);
      else if (which == 1)
        K[((size_t)(b * HW + row)) * C_DIM + d] = (_Float16)val;
      else
        V[((size_t)(b * C_DIM + d)) * HW + row] = (_Float16)val;
    }
  }
}

// ---------------------------------------------------------------------------
// Kernel 4: flash attention, TDM double-buffered K/V tiles in LDS.
// LDS: buf0 [0,64K) = K(32x512)+V(512x32); buf1 [64K,128K); P tiles at 128K.
// ---------------------------------------------------------------------------
#define KV_BUF_BYTES 65536u
#define P_OFF        131072u

__global__ void __launch_bounds__(256, 1)
attn_kernel(const _Float16* __restrict__ Q,
            const _Float16* __restrict__ K,
            const _Float16* __restrict__ V,
            _Float16* __restrict__ O) {
  extern __shared__ char smem[];

  int wave = threadIdx.x >> 5;
  int lane = threadIdx.x & 31;
  int lane15 = lane & 15;
  int halfId = lane >> 4;
  int tile = blockIdx.x * 8 + wave;
  int b  = tile >> 8;
  int q0 = (tile & 255) * 16;

  const _Float16* Kb = K + (size_t)b * HW * C_DIM;
  const _Float16* Vb = V + (size_t)b * C_DIM * HW;
  _Float16* Pp = (_Float16*)(smem + P_OFF) + wave * 512;   // [16][32] per wave

  // preload Q tile (16x512) into 16 A-fragments
  v16h qf[16];
  {
    const h8* Qrow = (const h8*)(Q + ((size_t)(b * HW + q0 + lane15)) * C_DIM);
    #pragma unroll
    for (int ks = 0; ks < 16; ++ks) {
      V16 a;
      a.h[0] = Qrow[ks * 4 + halfId];
      a.h[1] = Qrow[ks * 4 + halfId + 2];
      qf[ks] = a.v;
    }
  }

  v8f o_acc[32];
  #pragma unroll
  for (int j = 0; j < 32; ++j) o_acc[j] = v8f_zero();
  float mrow[8], lrow[8];
  #pragma unroll
  for (int r = 0; r < 8; ++r) { mrow[r] = -3.0e38f; lrow[r] = 0.0f; }

  // prologue: TDM-prefetch tile 0 into buffer 0 (wave 0 only; EXEC-independent)
  if (threadIdx.x < 32) {
    tdm_load_2d_f16(0u, Kb, C_DIM, 32, C_DIM);                    // K: 32x512
    tdm_load_2d_f16(32768u, Vb, 32, C_DIM, HW);                   // V: 512x32
  }

  const int NIT = HW / 32;                                        // 128
  for (int it = 0; it < NIT; ++it) {
    int sel = it & 1;
    const _Float16* Ks = (const _Float16*)(smem + sel * KV_BUF_BYTES);
    const _Float16* Vs = Ks + 32 * C_DIM;

    if (threadIdx.x < 32) __builtin_amdgcn_s_wait_tensorcnt(0);
    __syncthreads();                       // tile `it` resident for all waves

    // prefetch tile it+1 into the other buffer while computing on this one
    if (threadIdx.x < 32 && (it + 1) < NIT) {
      int kt1 = (it + 1) * 32;
      unsigned dst = (unsigned)((sel ^ 1) * KV_BUF_BYTES);
      tdm_load_2d_f16(dst, Kb + (size_t)kt1 * C_DIM, C_DIM, 32, C_DIM);
      tdm_load_2d_f16(dst + 32768u, Vb + kt1, 32, C_DIM, HW);
    }

    // ---- scores S[16x32] = Q * K^T (K B-fragments from LDS) ----
    v8f s0 = v8f_zero(), s1 = v8f_zero();
    const h8* Kr0 = (const h8*)(Ks + (size_t)lane15 * C_DIM);
    const h8* Kr1 = (const h8*)(Ks + (size_t)(16 + lane15) * C_DIM);
    #pragma unroll
    for (int ks = 0; ks < 16; ++ks) {
      V16 b0, b1;
      b0.h[0] = Kr0[ks * 4 + halfId];
      b0.h[1] = Kr0[ks * 4 + halfId + 2];
      b1.h[0] = Kr1[ks * 4 + halfId];
      b1.h[1] = Kr1[ks * 4 + halfId + 2];
      s0 = wmma_f16(qf[ks], b0.v, s0);
      s1 = wmma_f16(qf[ks], b1.v, s1);
    }

    // ---- online softmax ----
    float corr[8];
    #pragma unroll
    for (int r = 0; r < 8; ++r) {
      float v0 = s0[r], v1 = s1[r];
      float mx = fmaxf(v0, v1);
      mx = fmaxf(mx, __shfl_xor(mx, 1, 16));
      mx = fmaxf(mx, __shfl_xor(mx, 2, 16));
      mx = fmaxf(mx, __shfl_xor(mx, 4, 16));
      mx = fmaxf(mx, __shfl_xor(mx, 8, 16));
      float mnew = fmaxf(mrow[r], mx);
      float cr = __expf(mrow[r] - mnew);
      float p0 = __expf(v0 - mnew);
      float p1 = __expf(v1 - mnew);
      float sum = p0 + p1;
      sum += __shfl_xor(sum, 1, 16);
      sum += __shfl_xor(sum, 2, 16);
      sum += __shfl_xor(sum, 4, 16);
      sum += __shfl_xor(sum, 8, 16);
      lrow[r] = lrow[r] * cr + sum;
      mrow[r] = mnew;
      corr[r] = cr;
      int prow = r + 8 * halfId;
      Pp[prow * 32 + lane15]      = (_Float16)p0;
      Pp[prow * 32 + lane15 + 16] = (_Float16)p1;
    }
    __syncthreads();

    V16 pf;                                  // P as 16x32 A fragment
    {
      const h8* Lr = (const h8*)(Pp + lane15 * 32);
      pf.h[0] = Lr[halfId];
      pf.h[1] = Lr[halfId + 2];
    }

    #pragma unroll
    for (int j = 0; j < 32; ++j) {
      #pragma unroll
      for (int r = 0; r < 8; ++r) o_acc[j][r] *= corr[r];
    }

    // ---- O += P * V (V B-fragments from LDS, channel-major rows) ----
    #pragma unroll
    for (int j = 0; j < 32; ++j) {
      const h8* Vr = (const h8*)(Vs + (size_t)(j * 16 + lane15) * 32);
      V16 bv;
      bv.h[0] = Vr[halfId];
      bv.h[1] = Vr[halfId + 2];
      o_acc[j] = wmma_f16(pf.v, bv.v, o_acc[j]);
    }
    __syncthreads();                // all waves done with tile `it` buffers
  }

  #pragma unroll
  for (int j = 0; j < 32; ++j) {
    int d = j * 16 + lane15;
    #pragma unroll
    for (int r = 0; r < 8; ++r) {
      int row = q0 + r + 8 * halfId;
      O[((size_t)(b * HW + row)) * C_DIM + d] = (_Float16)(o_acc[j][r] / lrow[r]);
    }
  }
}

// ---------------------------------------------------------------------------
// Kernel 5: out projection + bias + residual; weights TDM-staged in LDS.
// ---------------------------------------------------------------------------
__global__ void outproj_kernel(const _Float16* __restrict__ O,
                               const _Float16* __restrict__ woT,
                               const float* __restrict__ bo,
                               const float* __restrict__ x,
                               float* __restrict__ out) {
  extern __shared__ _Float16 Ws[];              // [64][512]
  int b = blockIdx.z;
  int wave = threadIdx.x >> 5;
  int lane = threadIdx.x & 31;
  int lane15 = lane & 15;
  int halfId = lane >> 4;
  int m0 = blockIdx.y * 128 + wave * 16;
  int n0 = blockIdx.x * 64;

  if (threadIdx.x < 32) {
    tdm_load_2d_f16(0u, woT + (size_t)n0 * C_DIM, C_DIM, 64, C_DIM);
    __builtin_amdgcn_s_wait_tensorcnt(0);
  }
  __syncthreads();

  const h8* Arow = (const h8*)(O + ((size_t)(b * HW + m0 + lane15)) * C_DIM);
  v8f acc[4];
  #pragma unroll
  for (int j = 0; j < 4; ++j) acc[j] = v8f_zero();

  #pragma unroll
  for (int ks = 0; ks < 16; ++ks) {
    V16 a;
    a.h[0] = Arow[ks * 4 + halfId];
    a.h[1] = Arow[ks * 4 + halfId + 2];
    #pragma unroll
    for (int j = 0; j < 4; ++j) {
      const h8* Brow = (const h8*)(Ws + (size_t)(j * 16 + lane15) * C_DIM);
      V16 bb;
      bb.h[0] = Brow[ks * 4 + halfId];
      bb.h[1] = Brow[ks * 4 + halfId + 2];
      acc[j] = wmma_f16(a.v, bb.v, acc[j]);
    }
  }

  #pragma unroll
  for (int j = 0; j < 4; ++j) {
    int d = n0 + j * 16 + lane15;
    float bsv = bo[d];
    #pragma unroll
    for (int r = 0; r < 8; ++r) {
      int row = m0 + r + 8 * halfId;
      size_t idx = ((size_t)(b * C_DIM + d)) * HW + row;
      out[idx] = x[idx] + acc[j][r] + bsv;
    }
  }
}

// ---------------------------------------------------------------------------
extern "C" void kernel_launch(void* const* d_in, const int* in_sizes, int n_in,
                              void* d_out, int out_size, void* d_ws, size_t ws_size,
                              hipStream_t stream) {
  (void)in_sizes; (void)n_in; (void)out_size; (void)ws_size;
  const float* x   = (const float*)d_in[0];
  const float* gns = (const float*)d_in[1];
  const float* gnb = (const float*)d_in[2];
  const float* wq  = (const float*)d_in[3];
  const float* bq  = (const float*)d_in[4];
  const float* wk  = (const float*)d_in[5];
  const float* bk  = (const float*)d_in[6];
  const float* wv  = (const float*)d_in[7];
  const float* bv  = (const float*)d_in[8];
  const float* wo  = (const float*)d_in[9];
  const float* bo  = (const float*)d_in[10];
  float* out = (float*)d_out;

  char* ws = (char*)d_ws;
  const size_t szAct = (size_t)BATCH * HW * C_DIM * sizeof(_Float16);  // 16 MB
  const size_t szW   = (size_t)C_DIM * C_DIM * sizeof(_Float16);       // 512 KB
  _Float16* Xn  = (_Float16*)ws; ws += szAct;
  _Float16* Qm  = (_Float16*)ws; ws += szAct;
  _Float16* Km  = (_Float16*)ws; ws += szAct;
  _Float16* Vm  = (_Float16*)ws; ws += szAct;
  _Float16* Om  = (_Float16*)ws; ws += szAct;
  _Float16* wqT = (_Float16*)ws; ws += szW;
  _Float16* wkT = (_Float16*)ws; ws += szW;
  _Float16* wvT = (_Float16*)ws; ws += szW;
  _Float16* woT = (_Float16*)ws; ws += szW;

  prep_w_kernel<<<4096, 256, 0, stream>>>(wq, wk, wv, wo, wqT, wkT, wvT, woT);
  gn_kernel<<<dim3(NG, BATCH), 256, 0, stream>>>(x, gns, gnb, Xn);
  qkv_kernel<<<dim3(8, 32, 12), 256, 65536, stream>>>(Xn, wqT, wkT, wvT,
                                                      bq, bk, bv, Qm, Km, Vm);
  attn_kernel<<<dim3(128), 256, 139264, stream>>>(Qm, Km, Vm, Om);
  outproj_kernel<<<dim3(8, 32, 4), 256, 65536, stream>>>(Om, woT, bo, x, out);
}
